// BlockwiseQuantizationOptim_26345329394228
// MI455X (gfx1250) — matrix operand order repaired and checked by reading, not compiled, not run on hardware
//
#include <hip/hip_runtime.h>

// ---------------------------------------------------------------------------
// BlockwiseQuantizationOptim for MI455X (gfx1250, wave32)
//   - TDM tensor_load_to_lds stages each 128x128 f32 block (64 KB) into LDS
//   - v_wmma_f32_16x16x32_f16 reduces 256 softmax levels -> (sum e, sum e*q)
//   - windowed ds_add_f32 atomics build bin_mass[j][l] in LDS (128 KB)
//   - per-block entropy partial -> d_ws, tiny final kernel -> d_out[1024^2]
// ---------------------------------------------------------------------------

typedef __attribute__((ext_vector_type(16))) _Float16     v16h;
typedef __attribute__((ext_vector_type(8)))  float        v8f;
typedef __attribute__((ext_vector_type(4)))  unsigned int v4u;
typedef __attribute__((ext_vector_type(8)))  int          v8i;
typedef __attribute__((ext_vector_type(4)))  int          v4i;

#define BSZ   128
#define NLVL  256
#define EPSF  1e-6f
// exp(-T*|w_norm - l/255|) = exp2(-K2C*|u - l|), u = w_norm*255, T=100
#define K2C   0.5657618f        // (100/255)/ln2
#define LN2F  0.6931471806f

#if defined(__has_builtin)
#if __has_builtin(__builtin_amdgcn_tensor_load_to_lds)
#define HAVE_TDM 1
#endif
#endif

struct SMem {
  float tile[BSZ * BSZ];   //  64 KB, LDS offset 0 -> TDM destination
  float bin[BSZ * NLVL];   // 128 KB, bin_mass[j][l] (l contiguous: bank-clean atomics)
  float red[16];
};

__global__ __launch_bounds__(256)
void bq_main(const float* __restrict__ W,
             const float* __restrict__ wmin_p,
             const float* __restrict__ wmax_p,
             float* __restrict__ out,
             float* __restrict__ ws) {
  __shared__ SMem sm;

  const int tid  = threadIdx.x;
  const int lane = tid & 31;
  const int wave = tid >> 5;
  const int n    = blockIdx.x;          // block id 0..63
  const int bi   = n >> 3, bj = n & 7;
  const float* src = W + (size_t)(bi * BSZ) * 1024 + bj * BSZ;

#if HAVE_TDM
  if (tid < 32) {  // wave 0 issues the 2D tile DMA (EXEC ignored by TDM)
    unsigned long long ga = (unsigned long long)(const void*)src;
    v4u g0;
    g0[0] = 1u;                                   // count=1, user descriptor
    g0[1] = 0u;                                   // lds_addr = 0 (tile @ LDS base)
    g0[2] = (unsigned int)(ga & 0xFFFFFFFFull);   // global_addr[31:0]
    g0[3] = (unsigned int)((ga >> 32) & 0x1FFFFFFull) | (2u << 30); // addr hi | type=2
    v8i g1;
    g1[0] = 0x00020000;   // workgroup_mask=0, data_size=2 (4 B)
    g1[1] = 0x04000000;   // tensor_dim0 = 1024 (low 16 bits in [63:48])
    g1[2] = 0x04000000;   // tensor_dim0 hi=0 | tensor_dim1 = 1024
    g1[3] = 0x00800000;   // tensor_dim1 hi=0 | tile_dim0 = 128
    g1[4] = 0x00000080;   // tile_dim1 = 128 | tile_dim2 = 0
    g1[5] = 1024;         // tensor_dim0_stride = 1024 elements
    g1[6] = 0;
    g1[7] = 0;
    v4i gz4 = {0, 0, 0, 0};                       // D# groups 2/3: unused (2D tile)
    v8i gz8 = {0, 0, 0, 0, 0, 0, 0, 0};
    __builtin_amdgcn_tensor_load_to_lds(g0, g1, gz4, gz4, gz8, 0);
  }
#else
  for (int idx = tid; idx < BSZ * BSZ; idx += 256)
    sm.tile[idx] = src[(idx >> 7) * 1024 + (idx & 127)];
#endif

  // zero the bin_mass histogram while the DMA is in flight
  for (int idx = tid; idx < BSZ * NLVL; idx += 256) sm.bin[idx] = 0.0f;

#if HAVE_TDM
  if (tid < 32) __builtin_amdgcn_s_wait_tensorcnt(0);
#endif
  __syncthreads();

  // per-block clamp params (exactly the torch/jax clamping order)
  const float wmn   = wmin_p[n], wmx = wmax_p[n];
  const float lo    = fminf(wmn, wmx - EPSF);
  const float hi    = fmaxf(wmx, lo + EPSF);
  const float scale = hi - lo;
  const float u_k   = 255.0f / (scale + EPSF);

  // Constant B matrices (32x16 f16, 8 chunks of K=32 levels):
  //   column 0 = q_levels chunk (l/255), column 1 = ones, rest = 0.
  // B layout: lane = column N; K = 16*(vgpr>=4) + 8*(lane>=16) + 2*(vgpr&3) + half
  v16h bmat[8];
  {
    const int nn = lane & 15;
    const int kb = (lane >> 4) << 4;
#pragma unroll
    for (int kc = 0; kc < 8; ++kc) {
      v16h bb;
#pragma unroll
      for (int v = 0; v < 8; ++v) {
#pragma unroll
        for (int h = 0; h < 2; ++h) {
          const int l = kc * 32 + ((v >= 4) ? 16 : 0) + (kb >> 1) + (v & 3) * 2 + h;
          const float q = (float)l * (1.0f / 255.0f);
          const float val = (nn == 0) ? q : ((nn == 1) ? 1.0f : 0.0f);
          bb[v * 2 + h] = (_Float16)val;
        }
      }
      bmat[kc] = bb;
    }
  }

  // group = 16 consecutive elements of one row; 128 rows * 8 tiles = 1024 groups
  const int grp8 = (lane >> 4) << 3;   // A layout K offset for upper lane half
  for (int g = wave; g < BSZ * (BSZ / 16); g += 8) {
    const int i  = g >> 3;
    const int j0 = (g & 7) << 4;

    const float w = sm.tile[i * BSZ + j0 + (lane & 15)];
    const float u = (w - lo) * u_k;          // position in level units [0,255]

    v8f c = {0.f, 0.f, 0.f, 0.f, 0.f, 0.f, 0.f, 0.f};
#pragma unroll
    for (int kc = 0; kc < 8; ++kc) {
      // A (16x32 f16): lane holds element M=lane&15; K per ISA table:
      //   K = 16*(v>=4) + 8*(lane>=16) + 2*(v&3) + h
      v16h aa;
#pragma unroll
      for (int v = 0; v < 8; ++v) {
#pragma unroll
        for (int h = 0; h < 2; ++h) {
          const int   K  = ((v >= 4) ? 16 : 0) + grp8 + (v & 3) * 2 + h;
          const float lv = (float)(kc * 32 + K);
          aa[v * 2 + h] =
              (_Float16)__builtin_amdgcn_exp2f(-K2C * __builtin_fabsf(u - lv));
        }
      }
      c = __builtin_amdgcn_wmma_f32_16x16x32_f16(
          false, aa, false, bmat[kc], (short)0, c, false, false);
    }

    // D layout: column n = lane&15, row m = vgpr + 8*(lane>=16).
    // num = D[m,0] (lanes 0/16), den = D[m,1] (lanes 1/17) -> move to lane m.
    float num = 0.0f, den = 1.0f;
    const int srcA = (lane & 8) ? 16 : 0;
#pragma unroll
    for (int r = 0; r < 8; ++r) {
      const float nr = __shfl(c[r], srcA, 32);
      const float dr = __shfl(c[r], srcA + 1, 32);
      if ((lane & 7) == r) { num = nr; den = dr; }
    }
    const float invZ = 1.0f / den;           // meaningful on lanes 0..15

    if (lane < 16) {
      const float dq = num * invZ * scale + lo;    // w_q*scale + wmin
      out[(size_t)(bi * BSZ + i) * 1024 + bj * BSZ + j0 + lane] = dq;
    }

    // bin_mass: 64-level window around u (tail < 2^-18 of each element's mass).
    // 32 lanes cover consecutive l -> consecutive LDS banks, conflict-free.
#pragma unroll 4
    for (int m = 0; m < 16; ++m) {
      const float um = __shfl(u, m, 32);
      const float iz = __shfl(invZ, m, 32);
      int w0 = (int)um - 31;
      w0 = (w0 < 0) ? 0 : ((w0 > 192) ? 192 : w0);
      float* bjp = &sm.bin[(j0 + m) * NLVL];
#pragma unroll
      for (int k = 0; k < 2; ++k) {
        const int   l = w0 + k * 32 + lane;
        const float e = __builtin_amdgcn_exp2f(-K2C * __builtin_fabsf(um - (float)l));
        atomicAdd(&bjp[l], e * iz);
      }
    }
  }
  __syncthreads();

  // ---- per-block entropy: total mass, then -sum p*log(p+eps) ----
  float psum = 0.0f;
  for (int idx = tid; idx < BSZ * NLVL; idx += 256) psum += sm.bin[idx];
#pragma unroll
  for (int off = 16; off > 0; off >>= 1) psum += __shfl_down(psum, off, 32);
  if (lane == 0) sm.red[wave] = psum;
  __syncthreads();
  if (tid == 0) {
    float t = 0.0f;
    for (int k = 0; k < 8; ++k) t += sm.red[k];
    sm.red[8] = t;
  }
  __syncthreads();

  const float invT = 1.0f / (sm.red[8] + EPSF);
  float ep = 0.0f;
  for (int idx = tid; idx < BSZ * NLVL; idx += 256) {
    const float p = sm.bin[idx] * invT;
    ep -= p * (LN2F * __builtin_amdgcn_logf(p + EPSF));  // ln = log2 * ln2
  }
#pragma unroll
  for (int off = 16; off > 0; off >>= 1) ep += __shfl_down(ep, off, 32);
  __syncthreads();
  if (lane == 0) sm.red[wave] = ep;
  __syncthreads();
  if (tid == 0) {
    float t = 0.0f;
    for (int k = 0; k < 8; ++k) t += sm.red[k];
    ws[n] = t;
  }
}

__global__ void bq_final(const float* __restrict__ ws, float* __restrict__ out) {
  if (threadIdx.x == 0) {
    float s = 0.0f;
    for (int k = 0; k < 64; ++k) s += ws[k];
    out[1024u * 1024u] = s;   // entropy scalar appended after dequant
  }
}

extern "C" void kernel_launch(void* const* d_in, const int* in_sizes, int n_in,
                              void* d_out, int out_size, void* d_ws, size_t ws_size,
                              hipStream_t stream) {
  const float* W    = (const float*)d_in[0];
  const float* wmin = (const float*)d_in[1];
  const float* wmax = (const float*)d_in[2];
  float* out = (float*)d_out;
  float* ws  = (float*)d_ws;   // 64 floats of scratch

  bq_main<<<64, 256, 0, stream>>>(W, wmin, wmax, out, ws);
  bq_final<<<1, 32, 0, stream>>>(ws, out);
}
